// MultiInferRNNModel_55825984913740
// MI455X (gfx1250) — matrix operand order repaired and checked by reading, compiled, not run on hardware
//
#include <hip/hip_runtime.h>

// ---------------------------------------------------------------------------
// Types / WMMA helpers (CDNA5 gfx1250, wave32, v_wmma_f32_16x16x32_f16)
// ---------------------------------------------------------------------------
typedef _Float16 half_t;
typedef __attribute__((ext_vector_type(16))) _Float16 v16h;
typedef __attribute__((ext_vector_type(8)))  float    v8f;
typedef __attribute__((ext_vector_type(4)))  int      i32x4;

struct alignas(16) H8 { _Float16 h[8]; };

#if defined(__gfx1250__) &&                                                     \
    __has_builtin(__builtin_amdgcn_global_load_async_to_lds_b128) &&            \
    __has_builtin(__builtin_amdgcn_s_wait_asynccnt)
#define USE_ASYNC_LDS 1
#else
#define USE_ASYNC_LDS 0
#endif

__device__ __forceinline__ v8f wmma32(v16h a, v16h b, v8f c) {
  // D = A(16x32 f16) * B(32x16 f16) + C(16x16 f32)
  return __builtin_amdgcn_wmma_f32_16x16x32_f16(false, a, false, b, (short)0, c,
                                                false, false);
}

__device__ __forceinline__ v8f zero8() {
  v8f z = {0.f, 0.f, 0.f, 0.f, 0.f, 0.f, 0.f, 0.f};
  return z;
}

// A fragment (16x32, f16, row-major source, ld elements).
// lane<16: rows 0..15, K = k0+0..15
// lane>=16: same rows, K = k0+16..31
__device__ __forceinline__ v16h load_fragA(const _Float16* base, int ld, int row0,
                                           int k0, int lane) {
  const _Float16* p =
      base + (size_t)(row0 + (lane & 15)) * (size_t)ld + k0 + ((lane >> 4) << 4);
  H8 lo = *(const H8*)p;
  H8 hi = *(const H8*)(p + 8);
  v16h f;
#pragma unroll
  for (int i = 0; i < 8; ++i) { f[i] = lo.h[i]; f[8 + i] = hi.h[i]; }
  return f;
}

// 16 contiguous halfs -> fragment
__device__ __forceinline__ v16h load_frag_k16(const _Float16* p) {
  H8 a = *(const H8*)p;
  H8 b = *(const H8*)(p + 8);
  v16h f;
#pragma unroll
  for (int i = 0; i < 8; ++i) { f[i] = a.h[i]; f[8 + i] = b.h[i]; }
  return f;
}

// B fragment for C = A * W^T with W [N,K] row-major:
// lane -> col n = n0 + (lane&15); K = k0 + (lane>=16 ? 16 : 0) + i, i=0..15
__device__ __forceinline__ v16h load_fragB_nt(const _Float16* W, int ld, int n0,
                                              int k0, int lane) {
  return load_frag_k16(W + (size_t)(n0 + (lane & 15)) * (size_t)ld + k0 +
                       ((lane >> 4) << 4));
}

// B fragment for C = A * B with B [K,N] row-major (strided gather over K)
__device__ __forceinline__ v16h load_fragB_nn(const _Float16* Bm, int ld, int k0,
                                              int n0, int lane) {
  int n  = n0 + (lane & 15);
  int kb = k0 + ((lane >> 4) << 4);
  v16h f;
#pragma unroll
  for (int i = 0; i < 16; ++i) f[i] = Bm[(size_t)(kb + i) * (size_t)ld + n];
  return f;
}

__device__ __forceinline__ float sigm(float x) { return 1.f / (1.f + expf(-x)); }

// Copy 16 halfs (32B) global -> LDS; async (ASYNCcnt) on gfx1250 when available.
__device__ __forceinline__ void stage_copy(const half_t* g, _Float16* s) {
#if USE_ASYNC_LDS
  __builtin_amdgcn_global_load_async_to_lds_b128(
      (__attribute__((address_space(1))) i32x4*)(void*)g,
      (__attribute__((address_space(3))) i32x4*)(void*)s, 0, 0);
  __builtin_amdgcn_global_load_async_to_lds_b128(
      (__attribute__((address_space(1))) i32x4*)(void*)(g + 8),
      (__attribute__((address_space(3))) i32x4*)(void*)(s + 8), 0, 0);
#else
  ((H8*)s)[0] = ((const H8*)g)[0];
  ((H8*)s)[1] = ((const H8*)g)[1];
#endif
}
__device__ __forceinline__ void stage_wait_pipe() {  // next stage still in flight
#if USE_ASYNC_LDS
  __builtin_amdgcn_s_wait_asynccnt(4);
#endif
}
__device__ __forceinline__ void stage_wait_all() {
#if USE_ASYNC_LDS
  __builtin_amdgcn_s_wait_asynccnt(0);
#endif
}

// ---------------------------------------------------------------------------
// Generic per-wave GEMM:  C[M,N] = A[M,K] * W[N,K]^T (+bias) (+residual)
// f16 inputs, f32 accumulate.  Optional f32 and/or f16 outputs.
// Batched over blockIdx.y via element strides. blockDim.x == 128 (4 waves).
// M,N multiples of 16; K multiple of 32; stores limited to col < Nstore.
// ---------------------------------------------------------------------------
__global__ void gemm_nt_wave(const half_t* __restrict__ A, int lda, long long sA,
                             const half_t* __restrict__ W, int ldw, long long sW,
                             float* __restrict__ C, int ldc, long long sC,
                             half_t* __restrict__ Ch, int ldch, long long sCh,
                             const float* __restrict__ bias,
                             const float* __restrict__ resid, int ldr, long long sR,
                             int M, int N, int K, int Nstore) {
  int lane   = threadIdx.x & 31;
  int wave   = blockIdx.x * (blockDim.x >> 5) + (threadIdx.x >> 5);
  int tilesN = N >> 4;
  int tm = wave / tilesN, tn = wave % tilesN;
  if (tm >= (M >> 4)) return;
  int bb = blockIdx.y;
  const half_t* Ab = A + (size_t)bb * sA;
  const half_t* Wb = W + (size_t)bb * sW;
  v8f acc = zero8();
  for (int k0 = 0; k0 < K; k0 += 32) {
    v16h a = load_fragA(Ab, lda, tm * 16, k0, lane);
    v16h w = load_fragB_nt(Wb, ldw, tn * 16, k0, lane);
    acc = wmma32(a, w, acc);
  }
  int col   = tn * 16 + (lane & 15);
  int rbase = tm * 16 + ((lane >> 4) << 3);
  if (col < Nstore) {
    float bv = bias ? bias[col] : 0.f;
#pragma unroll
    for (int j = 0; j < 8; ++j) {
      int r   = rbase + j;
      float v = acc[j] + bv;
      if (resid) v += resid[(size_t)bb * sR + (size_t)r * ldr + col];
      if (C)  C [(size_t)bb * sC  + (size_t)r * ldc  + col] = v;
      if (Ch) Ch[(size_t)bb * sCh + (size_t)r * ldch + col] = (half_t)v;
    }
  }
}

// C[M,N] = A[M,K] * B[K,N] (+residual), same conventions as above.
__global__ void gemm_nn_wave(const half_t* __restrict__ A, int lda, long long sA,
                             const half_t* __restrict__ Bm, int ldb, long long sB,
                             float* __restrict__ C, int ldc, long long sC,
                             half_t* __restrict__ Ch, int ldch, long long sCh,
                             const float* __restrict__ resid, int ldr, long long sR,
                             int M, int N, int K) {
  int lane   = threadIdx.x & 31;
  int wave   = blockIdx.x * (blockDim.x >> 5) + (threadIdx.x >> 5);
  int tilesN = N >> 4;
  int tm = wave / tilesN, tn = wave % tilesN;
  if (tm >= (M >> 4)) return;
  int bb = blockIdx.y;
  const half_t* Ab = A  + (size_t)bb * sA;
  const half_t* Bb = Bm + (size_t)bb * sB;
  v8f acc = zero8();
  for (int k0 = 0; k0 < K; k0 += 32) {
    v16h a = load_fragA(Ab, lda, tm * 16, k0, lane);
    v16h w = load_fragB_nn(Bb, ldb, k0, tn * 16, lane);
    acc = wmma32(a, w, acc);
  }
  int col   = tn * 16 + (lane & 15);
  int rbase = tm * 16 + ((lane >> 4) << 3);
#pragma unroll
  for (int j = 0; j < 8; ++j) {
    int r   = rbase + j;
    float v = acc[j];
    if (resid) v += resid[(size_t)bb * sR + (size_t)r * ldr + col];
    if (C)  C [(size_t)bb * sC  + (size_t)r * ldc  + col] = v;
    if (Ch) Ch[(size_t)bb * sCh + (size_t)r * ldch + col] = (half_t)v;
  }
}

// ---------------------------------------------------------------------------
// LDS-tiled big GEMM: Ch[M,N](f16) = A[M,K] * W[N,K]^T + bias.
// 256 threads = 8 waves; 128x128 C tile per workgroup; 2-stage double-buffered
// software pipeline with async global->LDS tile staging (ASYNCcnt) on gfx1250.
// Wave (wm,wn), wm in 0..1, wn in 0..3, computes 4x2 16x16 subtiles.
// ---------------------------------------------------------------------------
#define ASTR 48  // 32 halfs data + 16 pad; 96B rows keep 16B alignment
__global__ void gemm_nt_tiled(const half_t* __restrict__ A, int lda,
                              const half_t* __restrict__ W, int ldw,
                              const float* __restrict__ bias,
                              half_t* __restrict__ Ch, int ldch,
                              int M, int N, int K) {
  __shared__ _Float16 As[2][128 * ASTR];
  __shared__ _Float16 Ws[2][128 * ASTR];
  int tid = threadIdx.x;
  int lane = tid & 31, wave = tid >> 5;
  int nBlocksN = N / 128;
  int bm = blockIdx.x / nBlocksN, bn = blockIdx.x % nBlocksN;
  int row0 = bm * 128, col0 = bn * 128;
  int wm = wave >> 2, wn = wave & 3;
  v8f acc[4][2];
#pragma unroll
  for (int i = 0; i < 4; ++i)
#pragma unroll
    for (int j = 0; j < 2; ++j) acc[i][j] = zero8();

  int lrow = tid >> 1;
  int lcol = (tid & 1) * 16;
  const half_t* gA = A + (size_t)(row0 + lrow) * lda + lcol;
  const half_t* gW = W + (size_t)(col0 + lrow) * ldw + lcol;
  _Float16* sAbase = &As[0][lrow * ASTR + lcol];
  _Float16* sWbase = &Ws[0][lrow * ASTR + lcol];
  const int bufStride = 128 * ASTR;

  int nk = K / 32;
  // stage 0
  stage_copy(gA, sAbase);
  stage_copy(gW, sWbase);
  for (int kc = 0; kc < nk; ++kc) {
    int cur = kc & 1;
    if (kc + 1 < nk) {  // prefetch next stage into the other buffer
      int nb = (kc + 1) & 1;
      stage_copy(gA + (size_t)(kc + 1) * 32, sAbase + nb * bufStride);
      stage_copy(gW + (size_t)(kc + 1) * 32, sWbase + nb * bufStride);
      stage_wait_pipe();   // own stage-kc loads complete (in-order), 4 in flight
    } else {
      stage_wait_all();
    }
    __syncthreads();       // all waves' tile data visible in LDS
    v16h af[4], wf[2];
#pragma unroll
    for (int i = 0; i < 4; ++i)
      af[i] = load_fragA(As[cur], ASTR, wm * 64 + i * 16, 0, lane);
#pragma unroll
    for (int j = 0; j < 2; ++j)
      wf[j] = load_fragB_nt(Ws[cur], ASTR, wn * 32 + j * 16, 0, lane);
#pragma unroll
    for (int i = 0; i < 4; ++i)
#pragma unroll
      for (int j = 0; j < 2; ++j) acc[i][j] = wmma32(af[i], wf[j], acc[i][j]);
    __syncthreads();       // done reading buf[cur] before it is refilled
  }
  int cl = lane & 15;
  int rh = (lane >> 4) << 3;
#pragma unroll
  for (int i = 0; i < 4; ++i)
#pragma unroll
    for (int j = 0; j < 2; ++j) {
      int col  = col0 + wn * 32 + j * 16 + cl;
      float bv = bias ? bias[col] : 0.f;
#pragma unroll
      for (int jj = 0; jj < 8; ++jj) {
        int row = row0 + wm * 64 + i * 16 + rh + jj;
        Ch[(size_t)row * ldch + col] = (half_t)(acc[i][j][jj] + bv);
      }
    }
}

// ---------------------------------------------------------------------------
// BiLSTM recurrence.  One 1024-thread workgroup per direction (gridDim.x=2).
// Per step: gates[16pad x 1024] = Hpad[16x256] @ whh[1024,256]^T + X[t] (WMMA),
// then elementwise LSTM cell.  h (f16) / c (f32) live in LDS.
// ctx layout: [b*128+t][512], forward -> cols 0..255, backward -> 256..511.
// ---------------------------------------------------------------------------
__global__ void lstm_kernel(const float* __restrict__ Xf, const float* __restrict__ Xb,
                            const half_t* __restrict__ whhF,
                            const half_t* __restrict__ whhB,
                            float* __restrict__ ctx) {
  int dir = blockIdx.x;
  const float*  X    = dir ? Xb : Xf;
  const half_t* whh  = dir ? whhB : whhF;
  int colOff = dir ? 256 : 0;

  __shared__ _Float16 hsh[16 * 256];
  __shared__ float    csh[4 * 256];
  __shared__ float    gbuf[4 * 1024];

  int tid = threadIdx.x, lane = tid & 31, wave = tid >> 5;  // 32 waves
  for (int i = tid; i < 16 * 256; i += 1024) hsh[i] = (half_t)0.f;
  for (int i = tid; i < 4 * 256;  i += 1024) csh[i] = 0.f;
  __syncthreads();

  for (int s = 0; s < 128; ++s) {
    int t = dir ? (127 - s) : s;
    v16h af[8];
#pragma unroll
    for (int kk = 0; kk < 8; ++kk) af[kk] = load_fragA(hsh, 256, 0, kk * 32, lane);
#pragma unroll
    for (int q = 0; q < 2; ++q) {
      int nt = wave * 2 + q;
      v8f acc = zero8();
#pragma unroll
      for (int kk = 0; kk < 8; ++kk) {
        v16h w = load_fragB_nt(whh, 256, nt * 16, kk * 32, lane);
        acc = wmma32(af[kk], w, acc);
      }
      if (lane < 16) {
        int col = nt * 16 + lane;
#pragma unroll
        for (int j = 0; j < 4; ++j)  // rows 0..3 == batches 0..3
          gbuf[j * 1024 + col] = acc[j] + X[(size_t)(j * 128 + t) * 1024 + col];
      }
    }
    __syncthreads();
    // LSTM cell: gates order i,f,g,o (torch layout)
    int b = tid >> 8, u = tid & 255;
    float ig = gbuf[b * 1024 + u];
    float fg = gbuf[b * 1024 + 256 + u];
    float gg = gbuf[b * 1024 + 512 + u];
    float og = gbuf[b * 1024 + 768 + u];
    float cn = sigm(fg) * csh[b * 256 + u] + sigm(ig) * tanhf(gg);
    float hn = sigm(og) * tanhf(cn);
    csh[b * 256 + u] = cn;
    hsh[b * 256 + u] = (half_t)hn;
    ctx[(size_t)(b * 128 + t) * 512 + colOff + u] = hn;
    __syncthreads();
  }
}

// ---------------------------------------------------------------------------
// Small helper kernels
// ---------------------------------------------------------------------------
__global__ void convert_pad(const float* __restrict__ src, half_t* __restrict__ dst,
                            int Rsrc, int Ksrc, int Kpad, int total) {
  int idx = blockIdx.x * blockDim.x + threadIdx.x;
  if (idx >= total) return;
  int k = idx % Kpad, r = idx / Kpad;
  float v = (r < Rsrc && k < Ksrc) ? src[(size_t)r * Ksrc + k] : 0.f;
  dst[idx] = (half_t)v;
}

__global__ void vec_add(const float* a, const float* b, float* o, int n) {
  int i = blockIdx.x * blockDim.x + threadIdx.x;
  if (i < n) o[i] = a[i] + b[i];
}

// Wlm[c][n] = feat_w[n][1024+c], c in 0..17 (lm_i | lm_j | probs blocks)
__global__ void wlm_build(const float* __restrict__ feat_w, float* __restrict__ Wlm) {
  int idx = blockIdx.x * blockDim.x + threadIdx.x;
  if (idx >= 18 * 1024) return;
  int n = idx & 1023, c = idx >> 10;
  Wlm[idx] = feat_w[(size_t)n * 1042 + 1024 + c];
}

__global__ void emb_gather(const int* __restrict__ tokens, const float* __restrict__ mask,
                           const float* __restrict__ gen, const float* __restrict__ dom,
                           half_t* __restrict__ emb) {
  int idx = blockIdx.x * blockDim.x + threadIdx.x;  // 512 * 416
  if (idx >= 512 * 416) return;
  int k = idx % 416, r = idx / 416;
  float v = 0.f;
  if (k < 400) {
    int tok = tokens[r];
    v = (k < 300) ? gen[(size_t)tok * 300 + k] : dom[(size_t)tok * 100 + (k - 300)];
    v *= mask[r];
  }
  emb[idx] = (half_t)v;
}

__global__ void softmax_kernel(const float* __restrict__ scores,
                               const float* __restrict__ mask,
                               half_t* __restrict__ attn) {
  int row = blockIdx.x;  // b*128+i, 512 rows
  int j   = threadIdx.x; // 0..127
  int b   = row >> 7;
  __shared__ float buf[128];
  float s = scores[(size_t)row * 128 + j];
  s = (mask[b * 128 + j] > 0.f) ? s : -1e30f;
  buf[j] = s;
  __syncthreads();
  for (int off = 64; off > 0; off >>= 1) {
    if (j < off) buf[j] = fmaxf(buf[j], buf[j + off]);
    __syncthreads();
  }
  float mx = buf[0];
  __syncthreads();
  float e = expf(s - mx);
  buf[j] = e;
  __syncthreads();
  for (int off = 64; off > 0; off >>= 1) {
    if (j < off) buf[j] += buf[j + off];
    __syncthreads();
  }
  attn[(size_t)row * 128 + j] = (half_t)(e / buf[0]);
}

// logits0[b,i,j,c] = La[b,i,c] + Lb[b,j,c]  (cls_b already folded into Lb)
__global__ void logits0_kernel(const float* __restrict__ La, const float* __restrict__ Lb,
                               float* __restrict__ out) {
  int idx = blockIdx.x * blockDim.x + threadIdx.x;  // 65536*6
  if (idx >= 65536 * 6) return;
  int c = idx % 6, r = idx / 6;
  int b = r >> 14, i = (r >> 7) & 127, j = r & 127;
  out[idx] = La[(size_t)(b * 128 + i) * 6 + c] + Lb[(size_t)(b * 128 + j) * 6 + c];
}

// lm[b,t,c] = max( max_i (P*m2)[b,i,t,c], max_j (P*m2)[b,t,j,c] ), m2 = triu(mask_i*mask_j)
__global__ void lm_reduce(const float* __restrict__ P, const float* __restrict__ mask,
                          float* __restrict__ lm) {
  int idx = blockIdx.x * blockDim.x + threadIdx.x;  // 4*128*6
  if (idx >= 4 * 128 * 6) return;
  int c = idx % 6, t = (idx / 6) % 128, b = idx / (6 * 128);
  const float* Pb = P + (size_t)b * 128 * 128 * 6;
  float mt = mask[b * 128 + t];
  float mmax = -3.4e38f;
  for (int i = 0; i < 128; ++i) {
    float m2 = (i <= t) ? mask[b * 128 + i] * mt : 0.f;
    mmax = fmaxf(mmax, Pb[((size_t)i * 128 + t) * 6 + c] * m2);
  }
  for (int j = 0; j < 128; ++j) {
    float m2 = (t <= j) ? mt * mask[b * 128 + j] : 0.f;
    mmax = fmaxf(mmax, Pb[((size_t)t * 128 + j) * 6 + c] * m2);
  }
  lm[(size_t)(b * 128 + t) * 6 + c] = mmax;
}

// Qa = Pa + lm @ Wlm[0:6]^T ;  Qb = Pb + lm @ Wlm[6:12]^T + feat_b  (in place)
__global__ void q_update(float* __restrict__ Pa, float* __restrict__ Pb,
                         const float* __restrict__ lm, const float* __restrict__ Wlm,
                         const float* __restrict__ feat_b) {
  int idx = blockIdx.x * blockDim.x + threadIdx.x;  // 512*1024
  if (idx >= 512 * 1024) return;
  int n = idx & 1023, r = idx >> 10;
  float qa = Pa[idx], qb = Pb[idx];
#pragma unroll
  for (int c = 0; c < 6; ++c) {
    float l = lm[(size_t)r * 6 + c];
    qa += l * Wlm[c * 1024 + n];
    qb += l * Wlm[(6 + c) * 1024 + n];
  }
  qb += feat_b[n];
  Pa[idx] = qa;
  Pb[idx] = qb;
}

// Hop-1 features, written straight into newfeat[:, 0:1024] (ld=1056); pad cols zeroed.
__global__ void hop1_build(const float* __restrict__ Pa, const float* __restrict__ Pb,
                           const float* __restrict__ P0, const float* __restrict__ Wlm,
                           half_t* __restrict__ newfeat) {
  int r = blockIdx.x;  // 65536
  int b = r >> 14, i = (r >> 7) & 127, j = r & 127;
  int rowI = b * 128 + i, rowJ = b * 128 + j;
  float p[6];
#pragma unroll
  for (int c = 0; c < 6; ++c) p[c] = P0[(size_t)r * 6 + c];
  for (int n = threadIdx.x; n < 1024; n += blockDim.x) {
    float v = Pa[(size_t)rowI * 1024 + n] + Pb[(size_t)rowJ * 1024 + n];
#pragma unroll
    for (int c = 0; c < 6; ++c) v += p[c] * Wlm[(12 + c) * 1024 + n];
    newfeat[(size_t)r * 1056 + n] = (half_t)v;
  }
  if (threadIdx.x < 14) newfeat[(size_t)r * 1056 + 1042 + threadIdx.x] = (half_t)0.f;
}

// Fill newfeat cols 1024..1041 with [lm_i(6) | lm_j(6) | probs(6)]
__global__ void fill_cols(const float* __restrict__ lm, const float* __restrict__ P1,
                          half_t* __restrict__ newfeat) {
  int idx = blockIdx.x * blockDim.x + threadIdx.x;  // 65536*18
  if (idx >= 65536 * 18) return;
  int c = idx % 18, r = idx / 18;
  int b = r >> 14, i = (r >> 7) & 127, j = r & 127;
  float v;
  if (c < 6)       v = lm[(size_t)(b * 128 + i) * 6 + c];
  else if (c < 12) v = lm[(size_t)(b * 128 + j) * 6 + (c - 6)];
  else             v = P1[(size_t)r * 6 + (c - 12)];
  newfeat[(size_t)r * 1056 + 1024 + c] = (half_t)v;
}

// ---------------------------------------------------------------------------
// Host launcher
// ---------------------------------------------------------------------------
#define CDIV(a, b) (((a) + (b) - 1) / (b))

extern "C" void kernel_launch(void* const* d_in, const int* in_sizes, int n_in,
                              void* d_out, int out_size, void* d_ws, size_t ws_size,
                              hipStream_t stream) {
  (void)in_sizes; (void)n_in; (void)out_size; (void)ws_size;

  const int*   tokens = (const int*)  d_in[0];
  const float* mask   = (const float*)d_in[2];
  const float* gen    = (const float*)d_in[3];
  const float* dom    = (const float*)d_in[4];
  const float* wih_f  = (const float*)d_in[5];
  const float* whh_f  = (const float*)d_in[6];
  const float* bih_f  = (const float*)d_in[7];
  const float* bhh_f  = (const float*)d_in[8];
  const float* wih_b  = (const float*)d_in[9];
  const float* whh_b  = (const float*)d_in[10];
  const float* bih_b  = (const float*)d_in[11];
  const float* bhh_b  = (const float*)d_in[12];
  const float* wq     = (const float*)d_in[13];
  const float* bq     = (const float*)d_in[14];
  const float* wk     = (const float*)d_in[15];
  const float* bk     = (const float*)d_in[16];
  const float* feat_w = (const float*)d_in[17];
  const float* feat_b = (const float*)d_in[18];
  const float* cls_w  = (const float*)d_in[19];
  const float* cls_b  = (const float*)d_in[20];
  float* out = (float*)d_out;

  // bump allocator over d_ws
  size_t off = 0;
  auto alloc = [&](size_t bytes) -> void* {
    void* p = (char*)d_ws + off;
    off += (bytes + 255) & ~(size_t)255;
    return p;
  };

  half_t* emb_h   = (half_t*)alloc((size_t)512 * 416 * 2);
  half_t* wihF_h  = (half_t*)alloc((size_t)1024 * 416 * 2);
  half_t* wihB_h  = (half_t*)alloc((size_t)1024 * 416 * 2);
  half_t* whhF_h  = (half_t*)alloc((size_t)1024 * 256 * 2);
  half_t* whhB_h  = (half_t*)alloc((size_t)1024 * 256 * 2);
  half_t* wq_h    = (half_t*)alloc((size_t)512 * 512 * 2);
  half_t* wk_h    = (half_t*)alloc((size_t)512 * 512 * 2);
  half_t* featw_h = (half_t*)alloc((size_t)1024 * 1056 * 2);
  half_t* clsw_h  = (half_t*)alloc((size_t)16 * 1024 * 2);
  float*  bsumF   = (float*) alloc((size_t)1024 * 4);
  float*  bsumB   = (float*) alloc((size_t)1024 * 4);
  float*  Xf      = (float*) alloc((size_t)512 * 1024 * 4);
  float*  Xb      = (float*) alloc((size_t)512 * 1024 * 4);
  float*  ctx     = (float*) alloc((size_t)512 * 512 * 4);
  half_t* ctx_h   = (half_t*)alloc((size_t)512 * 512 * 2);
  half_t* q_h     = (half_t*)alloc((size_t)512 * 512 * 2);
  half_t* k_h     = (half_t*)alloc((size_t)512 * 512 * 2);
  float*  scores  = (float*) alloc((size_t)4 * 128 * 128 * 4);
  half_t* attn_h  = (half_t*)alloc((size_t)4 * 128 * 128 * 2);
  float*  ctx2    = (float*) alloc((size_t)512 * 512 * 4);
  half_t* ctx2_h  = (half_t*)alloc((size_t)512 * 512 * 2);
  float*  Pa      = (float*) alloc((size_t)512 * 1024 * 4);
  float*  Pb      = (float*) alloc((size_t)512 * 1024 * 4);
  float*  La      = (float*) alloc((size_t)512 * 6 * 4);
  float*  Lb      = (float*) alloc((size_t)512 * 6 * 4);
  float*  logits0 = (float*) alloc((size_t)65536 * 6 * 4);
  float*  logits1 = (float*) alloc((size_t)65536 * 6 * 4);
  float*  lm      = (float*) alloc((size_t)512 * 6 * 4);
  float*  Wlm     = (float*) alloc((size_t)18 * 1024 * 4);
  half_t* newfeat = (half_t*)alloc((size_t)65536 * 1056 * 2);  // hop-2 input (hop-1 feats in cols 0..1023)
  half_t* feat2   = (half_t*)alloc((size_t)65536 * 1024 * 2);  // hop-2 output features

  const int T = 256;

  // ---- weight conversion / padding --------------------------------------
  convert_pad<<<CDIV(1024 * 416, T), T, 0, stream>>>(wih_f, wihF_h, 1024, 400, 416, 1024 * 416);
  convert_pad<<<CDIV(1024 * 416, T), T, 0, stream>>>(wih_b, wihB_h, 1024, 400, 416, 1024 * 416);
  convert_pad<<<CDIV(1024 * 256, T), T, 0, stream>>>(whh_f, whhF_h, 1024, 256, 256, 1024 * 256);
  convert_pad<<<CDIV(1024 * 256, T), T, 0, stream>>>(whh_b, whhB_h, 1024, 256, 256, 1024 * 256);
  convert_pad<<<CDIV(512 * 512, T), T, 0, stream>>>(wq, wq_h, 512, 512, 512, 512 * 512);
  convert_pad<<<CDIV(512 * 512, T), T, 0, stream>>>(wk, wk_h, 512, 512, 512, 512 * 512);
  convert_pad<<<CDIV(1024 * 1056, T), T, 0, stream>>>(feat_w, featw_h, 1024, 1042, 1056, 1024 * 1056);
  convert_pad<<<CDIV(16 * 1024, T), T, 0, stream>>>(cls_w, clsw_h, 6, 1024, 1024, 16 * 1024);
  vec_add<<<CDIV(1024, T), T, 0, stream>>>(bih_f, bhh_f, bsumF, 1024);
  vec_add<<<CDIV(1024, T), T, 0, stream>>>(bih_b, bhh_b, bsumB, 1024);
  wlm_build<<<CDIV(18 * 1024, T), T, 0, stream>>>(feat_w, Wlm);

  // ---- embedding gather --------------------------------------------------
  emb_gather<<<CDIV(512 * 416, T), T, 0, stream>>>(tokens, mask, gen, dom, emb_h);

  // ---- LSTM input projections: X = emb @ wih^T + (bih+bhh) ---------------
  gemm_nt_wave<<<dim3(512, 1), 128, 0, stream>>>(
      emb_h, 416, 0, wihF_h, 416, 0, Xf, 1024, 0, nullptr, 0, 0,
      bsumF, nullptr, 0, 0, 512, 1024, 416, 1024);
  gemm_nt_wave<<<dim3(512, 1), 128, 0, stream>>>(
      emb_h, 416, 0, wihB_h, 416, 0, Xb, 1024, 0, nullptr, 0, 0,
      bsumB, nullptr, 0, 0, 512, 1024, 416, 1024);

  // ---- BiLSTM recurrence (both directions in parallel) -------------------
  lstm_kernel<<<2, 1024, 0, stream>>>(Xf, Xb, whhF_h, whhB_h, ctx);
  convert_pad<<<CDIV(512 * 512, T), T, 0, stream>>>(ctx, ctx_h, 512, 512, 512, 512 * 512);

  // ---- attention: q, k ---------------------------------------------------
  gemm_nt_wave<<<dim3(256, 1), 128, 0, stream>>>(
      ctx_h, 512, 0, wq_h, 512, 0, nullptr, 0, 0, q_h, 512, 0,
      bq, nullptr, 0, 0, 512, 512, 512, 512);
  gemm_nt_wave<<<dim3(256, 1), 128, 0, stream>>>(
      ctx_h, 512, 0, wk_h, 512, 0, nullptr, 0, 0, k_h, 512, 0,
      bk, nullptr, 0, 0, 512, 512, 512, 512);

  // ---- scores[b] = q[b] @ k[b]^T  (batched) ------------------------------
  gemm_nt_wave<<<dim3(16, 4), 128, 0, stream>>>(
      q_h, 512, 128 * 512, k_h, 512, 128 * 512, scores, 128, 128 * 128,
      nullptr, 0, 0, nullptr, nullptr, 0, 0, 128, 128, 512, 128);

  // ---- softmax -----------------------------------------------------------
  softmax_kernel<<<512, 128, 0, stream>>>(scores, mask, attn_h);

  // ---- ctx2 = ctx + attn @ ctx  (batched) --------------------------------
  gemm_nn_wave<<<dim3(64, 4), 128, 0, stream>>>(
      attn_h, 128, 128 * 128, ctx_h, 512, 128 * 512, ctx2, 512, 128 * 512,
      ctx2_h, 512, 128 * 512, ctx, 512, 128 * 512, 128, 512, 128);

  // ---- hop-1 decomposed projections --------------------------------------
  gemm_nt_wave<<<dim3(512, 1), 128, 0, stream>>>(
      ctx2_h, 512, 0, featw_h, 1056, 0, Pa, 1024, 0, nullptr, 0, 0,
      nullptr, nullptr, 0, 0, 512, 1024, 512, 1024);
  gemm_nt_wave<<<dim3(512, 1), 128, 0, stream>>>(
      ctx2_h, 512, 0, featw_h + 512, 1056, 0, Pb, 1024, 0, nullptr, 0, 0,
      nullptr, nullptr, 0, 0, 512, 1024, 512, 1024);
  gemm_nt_wave<<<dim3(8, 1), 128, 0, stream>>>(
      ctx2_h, 512, 0, clsw_h, 1024, 0, La, 6, 0, nullptr, 0, 0,
      nullptr, nullptr, 0, 0, 512, 16, 512, 6);
  gemm_nt_wave<<<dim3(8, 1), 128, 0, stream>>>(
      ctx2_h, 512, 0, clsw_h + 512, 1024, 0, Lb, 6, 0, nullptr, 0, 0,
      cls_b, nullptr, 0, 0, 512, 16, 512, 6);

  // ---- logits0 -----------------------------------------------------------
  logits0_kernel<<<CDIV(65536 * 6, T), T, 0, stream>>>(La, Lb, logits0);

  // ---- hop 1 -------------------------------------------------------------
  lm_reduce<<<CDIV(4 * 128 * 6, T), T, 0, stream>>>(logits0, mask, lm);
  q_update<<<CDIV(512 * 1024, T), T, 0, stream>>>(Pa, Pb, lm, Wlm, feat_b);
  hop1_build<<<65536, 256, 0, stream>>>(Pa, Pb, logits0, Wlm, newfeat);
  gemm_nt_wave<<<dim3(1024, 1), 128, 0, stream>>>(
      newfeat, 1056, 0, clsw_h, 1024, 0, logits1, 6, 0, nullptr, 0, 0,
      cls_b, nullptr, 0, 0, 65536, 16, 1024, 6);

  // ---- hop 2 -------------------------------------------------------------
  lm_reduce<<<CDIV(4 * 128 * 6, T), T, 0, stream>>>(logits1, mask, lm);
  fill_cols<<<CDIV(65536 * 18, T), T, 0, stream>>>(lm, logits1, newfeat);
  // features2 = newfeat @ feat_w^T + feat_b : async double-buffered WMMA GEMM
  gemm_nt_tiled<<<dim3((65536 / 128) * (1024 / 128)), 256, 0, stream>>>(
      newfeat, 1056, featw_h, 1056, feat_b, feat2, 1024, 65536, 1024, 1056);
  // final logits -> d_out (f32 [4,128,128,6])
  gemm_nt_wave<<<dim3(1024, 1), 128, 0, stream>>>(
      feat2, 1024, 0, clsw_h, 1024, 0, out, 6, 0, nullptr, 0, 0,
      cls_b, nullptr, 0, 0, 65536, 16, 1024, 6);
}
